// ConvCNF_59390807769273
// MI455X (gfx1250) — matrix operand order repaired and checked by reading, not compiled, not run on hardware
//
#include <hip/hip_runtime.h>

typedef __attribute__((ext_vector_type(16))) __bf16 v16bf;
typedef __attribute__((ext_vector_type(8)))  float  v8f;

// Pack two f32 -> two bf16 in one dword (round-to-nearest, ties away from
// zero: differs from RNE only on exact-tie mantissas). v_add x2 + v_perm_b32.
__device__ __forceinline__ unsigned packbf(float a, float b) {
    unsigned ua = __float_as_uint(a) + 0x8000u;
    unsigned ub = __float_as_uint(b) + 0x8000u;
#if __has_builtin(__builtin_amdgcn_perm)
    // result bytes: {ub[3],ub[2],ua[3],ua[2]} -> low half = bf16(a), high = bf16(b)
    return __builtin_amdgcn_perm(ub, ua, 0x07060302u);
#else
    return (ua >> 16) | (ub & 0xFFFF0000u);
#endif
}

// ---------------------------------------------------------------------------
// Setup: t-embedding -> kp -> per-orbit feature weights W_o[50], divergence
// weights dW[50], and the bf16 B-matrix register image (2 K-chunks x 32 lanes
// x 8 dwords) for V_WMMA_F32_16X16X32_BF16.
// Dense B (32x16, KxN) layout: lanes 0-15 hold N=lane, K=0..7,16..23 (2 per VGPR);
// lanes 16-31 hold N=lane-16, K=8..15,24..31.
// ---------------------------------------------------------------------------
__global__ void cnf_setup(const float* __restrict__ tptr,
                          const float* __restrict__ time_red,     // (20,21)
                          const float* __restrict__ kparams,      // (3,20,20)
                          const float* __restrict__ fsuper,       // (20,50)
                          float* __restrict__ dWg,                // [50]
                          unsigned* __restrict__ Bpack)           // [512]
{
    __shared__ float base[21];
    __shared__ float temb[20];
    __shared__ float kp[60];
    __shared__ float W[150];
    const int tid = threadIdx.x;
    const float t = tptr[0];

    if (tid < 21) {
        float v;
        if (tid == 0)        v = 1.0f;
        else if (tid <= 10)  v = __cosf(6.2831853071795864f * (float)tid * t);
        else                 v = __sinf(6.2831853071795864f * (float)(tid - 10) * t);
        base[tid] = v;
    }
    __syncthreads();
    if (tid < 20) {
        float a = 0.f;
        for (int j = 0; j < 21; ++j) a = fmaf(time_red[tid * 21 + j], base[j], a);
        temb[tid] = a;
    }
    __syncthreads();
    if (tid < 60) {                                  // kp[o,i] = sum_t kparams[o,i,t]*temb[t]
        float a = 0.f;
        for (int j = 0; j < 20; ++j) a = fmaf(kparams[tid * 20 + j], temb[j], a);
        kp[tid] = a;
    }
    __syncthreads();
    for (int e = tid; e < 150; e += 64) {            // W_o[f] = sum_i kp[o,i]*fs[i,f]/50
        int o = e / 50, f = e % 50;
        float a = 0.f;
        for (int i = 0; i < 20; ++i) a = fmaf(kp[o * 20 + i], fsuper[i * 50 + f], a);
        W[e] = a * 0.02f;
    }
    __syncthreads();
    if (tid < 50) dWg[tid] = (tid < 49) ? (float)(tid + 1) * W[tid] : W[49];

    for (int e = tid; e < 512; e += 64) {
        int c = e >> 8, rem = e & 255, lane = rem >> 3, j = rem & 7;
        int n   = lane & 15;
        int Klo = c * 32 + ((lane < 16) ? 0 : 8) + ((j < 4) ? 2 * j : 8 + 2 * j);
        float w0 = (n < 3 && Klo     < 50) ? W[n * 50 + Klo]     : 0.f;
        float w1 = (n < 3 && Klo + 1 < 50) ? W[n * 50 + Klo + 1] : 0.f;
        Bpack[e] = packbf(w0, w1);
    }
}

// ---------------------------------------------------------------------------
// Pass 1: 1024 pixels per block (4 x 256), amortizing B fragments, pointer
// selects and the reduction over 4 pixels/thread. Per pixel v: Chebyshev
// 3-term recurrence generates sin(kv),cos(kv), k=1..49 at 1 FMA per value.
// Feature row (bf16, K=64): [sin(1v)..sin(49v), v, 0..0] staged in LDS via
// batched b128 stores. All 8 A-fragment ds_load_b128 are issued before the
// first of the 4 v_wmma_f32_16x16x32_bf16 so the waits drain while the
// matrix pipe runs. s0,s1,s2 go straight from D fragments to global.
// Divergence in VALU (SGPR weights), deterministic two-level reduction.
// ---------------------------------------------------------------------------
__global__ void __launch_bounds__(256) cnf_pass1(
    const float* __restrict__ x,
    const float* __restrict__ dWg,
    const unsigned* __restrict__ Bpack,
    float* __restrict__ S0,          // d_out grad region, used as scratch
    float* __restrict__ S1,
    float* __restrict__ S2,
    float* __restrict__ blockSums)
{
    __shared__ unsigned feats[256 * 36];   // 256 rows x 32 data dwords, stride 36 (16B aligned)
    __shared__ float red8[8];

    const int tid  = threadIdx.x;
    const int lane = tid & 31;
    const int wave = tid >> 5;

    // B operands: per-lane 32B slice per K-chunk (global_load_b128 x2 each)
    union BU { uint4 q[2]; v16bf v; } b0u, b1u;
    {
        const uint4* bq = (const uint4*)Bpack;
        b0u.q[0] = bq[lane * 2 + 0];
        b0u.q[1] = bq[lane * 2 + 1];
        b1u.q[0] = bq[(32 + lane) * 2 + 0];
        b1u.q[1] = bq[(32 + lane) * 2 + 1];
    }

    // Hoisted per-thread constants
    const int   blockBase = blockIdx.x * 1024;
    const int   n         = lane & 15;
    const bool  writer    = (n < 3);
    float*      Sp        = (n == 0) ? S0 : ((n == 1) ? S1 : S2);
    const int   sOff      = wave * 32 + ((lane < 16) ? 0 : 8);   // + g*16 per group
    const int   halfsel   = (lane < 16) ? 0 : 4;
    const unsigned aRow0  = (unsigned)(wave * 32 + n) * 36u;
    const unsigned aRow1  = aRow0 + 16u * 36u;
    uint4* rowq = (uint4*)&feats[(unsigned)tid * 36u];
    const float dW49 = dWg[49];
    float dtot = 0.f;

    #pragma unroll 1
    for (int it = 0; it < 4; ++it) {
        const int pbase = blockBase + it * 256;
        const float v = x[pbase + tid];

        float s1v, c1v;
        __sincosf(v, &s1v, &c1v);
        const float c2 = c1v + c1v;
        float sPrev = 0.f, sCur = s1v;        // sin(0v), sin(1v)
        float cPrev = 1.f, cCur = c1v;        // cos(0v), cos(1v)
        float dacc0 = dW49;
        float dacc1 = 0.f;

        unsigned w0 = 0, w1 = 0, w2 = 0, w3 = 0;
        #pragma unroll
        for (int d = 0; d < 24; ++d) {        // emits sin(2d+1), sin(2d+2)
            const float sA = sCur;
            dacc0 = fmaf(dWg[2 * d], cCur, dacc0);            // k = 2d+1 term
            float sN = fmaf(c2, sCur, -sPrev); sPrev = sCur; sCur = sN;
            float cN = fmaf(c2, cCur, -cPrev); cPrev = cCur; cCur = cN;
            const float sB = sCur;
            dacc1 = fmaf(dWg[2 * d + 1], cCur, dacc1);        // k = 2d+2 term
            sN = fmaf(c2, sCur, -sPrev); sPrev = sCur; sCur = sN;
            cN = fmaf(c2, cCur, -cPrev); cPrev = cCur; cCur = cN;
            const unsigned w = packbf(sA, sB);
            switch (d & 3) {
                case 0: w0 = w; break;
                case 1: w1 = w; break;
                case 2: w2 = w; break;
                default:
                    w3 = w;
                    rowq[d >> 2] = make_uint4(w0, w1, w2, w3);
            }
        }
        dacc0 = fmaf(dWg[48], cCur, dacc0);                   // k = 49 term
        dtot += dacc0 + dacc1;
        rowq[6] = make_uint4(packbf(sCur, v), 0u, 0u, 0u);    // sin(49v), poly feature v
        rowq[7] = make_uint4(0u, 0u, 0u, 0u);                 // K = 56..63 zero pad
        __syncthreads();                                      // cross-lane RAW on feats

        // A layout (16x32 bf16): lanes 0-15 K=0..7,16..23; lanes 16-31 K=8..15,24..31.
        // Prefetch all 4 fragment pairs (8 x ds_load_b128) before any WMMA.
        union AU { uint4 q[2]; v16bf v; } aG0K0, aG0K1, aG1K0, aG1K1;
        aG0K0.q[0] = *(const uint4*)&feats[aRow0 + halfsel + 0];
        aG0K0.q[1] = *(const uint4*)&feats[aRow0 + halfsel + 8];
        aG0K1.q[0] = *(const uint4*)&feats[aRow0 + 16 + halfsel + 0];
        aG0K1.q[1] = *(const uint4*)&feats[aRow0 + 16 + halfsel + 8];
        aG1K0.q[0] = *(const uint4*)&feats[aRow1 + halfsel + 0];
        aG1K0.q[1] = *(const uint4*)&feats[aRow1 + halfsel + 8];
        aG1K1.q[0] = *(const uint4*)&feats[aRow1 + 16 + halfsel + 0];
        aG1K1.q[1] = *(const uint4*)&feats[aRow1 + 16 + halfsel + 8];

        v8f acc0 = {};
        acc0 = __builtin_amdgcn_wmma_f32_16x16x32_bf16(false, aG0K0.v, false, b0u.v,
                                                       (short)0, acc0, false, false);
        acc0 = __builtin_amdgcn_wmma_f32_16x16x32_bf16(false, aG0K1.v, false, b1u.v,
                                                       (short)0, acc0, false, false);
        v8f acc1 = {};
        acc1 = __builtin_amdgcn_wmma_f32_16x16x32_bf16(false, aG1K0.v, false, b0u.v,
                                                       (short)0, acc1, false, false);
        acc1 = __builtin_amdgcn_wmma_f32_16x16x32_bf16(false, aG1K1.v, false, b1u.v,
                                                       (short)0, acc1, false, false);

        // D layout: VGPR r: lanes0-15 -> (M=r, N=lane); lanes16-31 -> (M=8+r, N=lane-16).
        // Rows map to consecutive pixels -> two 16B stores per holder lane per group.
        if (writer) {
            const int pix0 = pbase + sOff;
            *(float4*)(Sp + pix0)      = make_float4(acc0[0], acc0[1], acc0[2], acc0[3]);
            *(float4*)(Sp + pix0 + 4)  = make_float4(acc0[4], acc0[5], acc0[6], acc0[7]);
            *(float4*)(Sp + pix0 + 16) = make_float4(acc1[0], acc1[1], acc1[2], acc1[3]);
            *(float4*)(Sp + pix0 + 20) = make_float4(acc1[4], acc1[5], acc1[6], acc1[7]);
        }
        __syncthreads();                   // cross-iteration WAR on feats
    }

    // Deterministic reduction: wave32 butterfly, then 8-way scalar sum.
    float r = dtot;
    #pragma unroll
    for (int off = 16; off > 0; off >>= 1) r += __shfl_xor(r, off, 32);
    if (lane == 0) red8[wave] = r;
    __syncthreads();
    if (tid == 0) {
        float s = 0.f;
        #pragma unroll
        for (int i = 0; i < 8; ++i) s += red8[i];
        blockSums[blockIdx.x] = s;
    }
}

// ---------------------------------------------------------------------------
// Pass 2: 3x3 D4 stencil with periodic wrap. Center from S0 (in d_out),
// edges from S1, corners from S2. Reuse rides the 192MB L2.
// ---------------------------------------------------------------------------
__global__ void __launch_bounds__(256) cnf_pass2(float* __restrict__ grad,
                                                 const float* __restrict__ S1,
                                                 const float* __restrict__ S2)
{
    const int p  = blockIdx.x * 256 + threadIdx.x;
    const int w  = p & 127, h = (p >> 7) & 127;
    const int ib = p & ~16383;
    const int hu = (h + 127) & 127, hd = (h + 1) & 127;
    const int wl = (w + 127) & 127, wr = (w + 1) & 127;
    const int rC = ib + (h << 7), rU = ib + (hu << 7), rD = ib + (hd << 7);
    float vv = grad[p];
    vv += S1[rC + wl] + S1[rC + wr] + S1[rU + w]  + S1[rD + w];
    vv += S2[rU + wl] + S2[rU + wr] + S2[rD + wl] + S2[rD + wr];
    grad[p] = vv;
}

// Per-image deterministic reduction of 16 block partials; output -div.
__global__ void cnf_div(const float* __restrict__ blockSums,
                        float* __restrict__ divOut, int Bimg)
{
    const int b = threadIdx.x;
    if (b < Bimg) {
        float s = 0.f;
        for (int j = 0; j < 16; ++j) s += blockSums[b * 16 + j];
        divOut[b] = -s;
    }
}

extern "C" void kernel_launch(void* const* d_in, const int* in_sizes, int n_in,
                              void* d_out, int out_size, void* d_ws, size_t ws_size,
                              hipStream_t stream)
{
    const float* t        = (const float*)d_in[0];
    const float* x        = (const float*)d_in[1];
    const float* time_red = (const float*)d_in[2];
    const float* kparams  = (const float*)d_in[3];
    const float* fsuper   = (const float*)d_in[4];

    const int P     = in_sizes[1];    // B*128*128
    const int Bimg  = P >> 14;
    const int nblk1 = P >> 10;        // pass1: 1024 pixels/block
    const int nblk2 = P >> 8;         // pass2: 256 pixels/block

    float*    S1        = (float*)d_ws;
    float*    S2        = S1 + P;
    float*    dWg       = S2 + P;
    unsigned* Bpack     = (unsigned*)(dWg + 64);
    float*    blockSums = (float*)(Bpack + 512);

    float* grad   = (float*)d_out;    // P floats (also S0 scratch)
    float* divOut = grad + P;         // Bimg floats

    cnf_setup<<<1, 64, 0, stream>>>(t, time_red, kparams, fsuper, dWg, Bpack);
    cnf_pass1<<<nblk1, 256, 0, stream>>>(x, dWg, Bpack, grad, S1, S2, blockSums);
    cnf_pass2<<<nblk2, 256, 0, stream>>>(grad, S1, S2);
    cnf_div<<<1, 128, 0, stream>>>(blockSums, divOut, Bimg);
}
